// PatchGCN_varpool_27032524161724
// MI455X (gfx1250) — matrix operand order repaired and checked by reading, not compiled, not run on hardware
//
#include <hip/hip_runtime.h>
#include <stdint.h>

// ---------------- constants from the reference ----------------
#define HDIM   128
#define NLAY   3
#define DCAT   512
#define NVP    100
#define NCLS   4
#define FIN    1024
#define EPS_MSG 1e-7f
#define LN_EPS  1e-5f
#define LOG_EPS 0.01f

// ---------------- vector types for WMMA ----------------
typedef __attribute__((ext_vector_type(16))) __bf16 bf16x16;
typedef __attribute__((ext_vector_type(8)))  __bf16 bf16x8;
typedef __attribute__((ext_vector_type(8)))  float  f32x8;

__device__ inline bf16x16 cat8(bf16x8 a, bf16x8 b) {
  return __builtin_shufflevector(a, b, 0,1,2,3,4,5,6,7,8,9,10,11,12,13,14,15);
}

__device__ inline void atomAddF(float* p, float v) {
  __hip_atomic_fetch_add(p, v, __ATOMIC_RELAXED, __HIP_MEMORY_SCOPE_AGENT);
}
// IEEE-monotonic float atomic max (works for mixed signs, init to -inf)
__device__ inline void atomMaxF(float* p, float v) {
  if (v >= 0.0f) atomicMax((int*)p, __float_as_int(v));
  else           atomicMin((unsigned int*)p, (unsigned int)__float_as_int(v));
}

__device__ inline float apply_act(float x, int act) {
  switch (act) {
    case 1:  return fmaxf(x, 0.0f);
    case 2:  return tanhf(x);
    case 3:  return 1.0f / (1.0f + __expf(-x));
    default: return x;
  }
}

// =====================================================================
// Tiled bf16 WMMA GEMM:  C[M,Ncols] = epilogue(A[M,K] @ B[K,Ncols] + bias)
// Block tile 128(M) x 128(N), K-step 64. 8 waves in a 4x2 grid; each wave
// owns a 32x64 output tile = 8 accumulators; 16 v_wmma_f32_16x16x32_bf16
// per barrier (2 K-chunks x (2 A-frags x 4 B-frags)).
// LDS A: row-major [128][72] halves; LDS B: transposed [128][72] so each
// lane's fragment is two 16B reads (ISA 7.12.2 layouts).
// Out-of-range rows/cols are CLAMP-loaded (they only affect output
// elements that the store guard drops) -> no divergent branch in the loop.
// cmode==1: C *= epilogue(...)   (used for the sigmoid gate)
// Requires K % 64 == 0 (holds for all call sites: 1024/128/256/512).
// =====================================================================
#define GBM 128
#define GBN 128
#define GBK 64
#define LDS_P 72   // halves per row (144B pitch: 16B aligned, conflict-friendly)

__global__ __launch_bounds__(256)
void k_gemm(const float* __restrict__ A, int lda,
            const float* __restrict__ B, int ldb,
            const float* __restrict__ bias,
            float* __restrict__ C, int ldc,
            int M, int K, int Ncols, int act, int cmode)
{
  __shared__ __bf16 sA[GBM * LDS_P];
  __shared__ __bf16 sB[GBN * LDS_P];

  const int tid  = threadIdx.x;
  const int wid  = tid >> 5;
  const int lane = tid & 31;
  const int wrow = wid >> 1;      // 0..3 : 32-row slab
  const int wcol = wid & 1;       // 0..1 : 64-col slab
  const int m0 = blockIdx.x * GBM;
  const int n0 = blockIdx.y * GBN;

  f32x8 acc[2][4] = {};

  const int lrow = lane & 15;
  const int kb   = (lane < 16) ? 0 : 8;    // A fragment K-base (ISA layout)
  const int kh   = (lane < 16) ? 0 : 8;    // B fragment K-base

  // staging coordinates
  const int ar = tid >> 1;            // A row 0..127
  const int ac = (tid & 1) * 32;      // A col segment (32 floats)
  const int bk = tid >> 3;            // B k-row 0..31 within a 32-chunk
  const int bc = (tid & 7) * 16;      // B col segment (16 floats)

  for (int k0 = 0; k0 < K; k0 += GBK) {
    // ---- stage A tile (128 x 64) fp32 -> bf16, clamp-load OOB rows ----
    {
      const int gr = min(m0 + ar, M - 1);
      const float* src = A + (size_t)gr * lda + k0 + ac;
      __builtin_prefetch(src + GBK, 0, 1);   // global_prefetch_b8 next K tile
      __bf16* dst = &sA[(size_t)ar * LDS_P + ac];
      #pragma unroll
      for (int j = 0; j < 32; ++j) dst[j] = (__bf16)src[j];
    }
    // ---- stage B tile (64 x 128) transposed into sB[col][k], clamp cols ----
    #pragma unroll
    for (int kc = 0; kc < 2; ++kc) {
      const int kk = kc * 32 + bk;
      const float* src = B + (size_t)(k0 + kk) * ldb;
      #pragma unroll
      for (int j = 0; j < 16; ++j) {
        const int gc = min(n0 + bc + j, Ncols - 1);
        sB[(size_t)(bc + j) * LDS_P + kk] = (__bf16)src[gc];
      }
    }
    __syncthreads();

    // ---- 2 K-chunks x 8 WMMAs ----
    #pragma unroll
    for (int kc = 0; kc < 2; ++kc) {
      const int ko = kc * 32;
      const __bf16* pa0 = &sA[(size_t)(wrow * 32 + lrow) * LDS_P + ko + kb];
      const __bf16* pa1 = pa0 + 16 * LDS_P;
      bf16x16 a0 = cat8(*(const bf16x8*)pa0, *(const bf16x8*)(pa0 + 16));
      bf16x16 a1 = cat8(*(const bf16x8*)pa1, *(const bf16x8*)(pa1 + 16));
      #pragma unroll
      for (int tj = 0; tj < 4; ++tj) {
        const __bf16* pb = &sB[(size_t)(wcol * 64 + tj * 16 + lrow) * LDS_P + ko + kh];
        bf16x16 b = cat8(*(const bf16x8*)pb, *(const bf16x8*)(pb + 16));
        acc[0][tj] = __builtin_amdgcn_wmma_f32_16x16x32_bf16(false, a0, false, b, (short)0, acc[0][tj], false, false);
        acc[1][tj] = __builtin_amdgcn_wmma_f32_16x16x32_bf16(false, a1, false, b, (short)0, acc[1][tj], false, false);
      }
    }
    __syncthreads();
  }

  // ---- epilogue: C/D layout (VGPR v: lanes0-15 -> M=v, lanes16-31 -> M=v+8)
  #pragma unroll
  for (int ti = 0; ti < 2; ++ti) {
    #pragma unroll
    for (int tj = 0; tj < 4; ++tj) {
      const int gcol = n0 + wcol * 64 + tj * 16 + lrow;
      if (gcol >= Ncols) continue;
      const float bsv = bias ? bias[gcol] : 0.0f;
      #pragma unroll
      for (int v = 0; v < 8; ++v) {
        const int grow = m0 + wrow * 32 + ti * 16 + ((lane < 16) ? v : v + 8);
        if (grow >= M) continue;
        float val = apply_act(acc[ti][tj][v] + bsv, act);
        const size_t off = (size_t)grow * ldc + gcol;
        if (cmode == 1) C[off] *= val; else C[off] = val;
      }
    }
  }
}

// =====================================================================
// fill (by bit pattern)
// =====================================================================
__global__ void k_fill(uint32_t* p, uint32_t v, size_t n) {
  size_t i = (size_t)blockIdx.x * blockDim.x + threadIdx.x;
  if (i < n) p[i] = v;
}

// =====================================================================
// GENConv message passing, feature-parallel (128 threads per edge)
// mode 0: segment max of t*msg     mode 1: segment sum of exp(t*msg - mx)
// mode 2: segment sum of alpha*msg
// =====================================================================
__global__ void k_edge(const float* __restrict__ feats, int ldf, int colbase,
                       const int* __restrict__ src, const int* __restrict__ dst,
                       const float* __restrict__ tptr, int layer,
                       float* __restrict__ mx, float* __restrict__ den,
                       float* __restrict__ agg, int E, int mode)
{
  long long idx = (long long)blockIdx.x * blockDim.x + threadIdx.x;
  if (idx >= (long long)E * HDIM) return;
  const int e = (int)(idx >> 7);
  const int f = (int)(idx & (HDIM - 1));
  const int s = src[e], d = dst[e];
  const float h   = feats[(size_t)s * ldf + colbase + f];
  const float msg = fmaxf(h, 0.0f) + EPS_MSG;
  const float sm  = tptr[layer] * msg;
  const size_t o  = (size_t)d * HDIM + f;
  if (mode == 0) {
    atomMaxF(&mx[o], sm);
  } else {
    const float ex = __expf(sm - mx[o]);
    if (mode == 1) atomAddF(&den[o], ex);
    else           atomAddF(&agg[o], (ex / den[o]) * msg);
  }
}

// agg[n,f] += feats[n, colbase+f]   (root skip)
__global__ void k_add_root(float* __restrict__ agg, const float* __restrict__ feats,
                           int ldf, int colbase, int n) {
  long long idx = (long long)blockIdx.x * blockDim.x + threadIdx.x;
  if (idx >= (long long)n * HDIM) return;
  const int r = (int)(idx >> 7), f = (int)(idx & (HDIM - 1));
  agg[idx] += feats[(size_t)r * ldf + colbase + f];
}

// =====================================================================
// Row LayerNorm + ReLU, optional residual add.  blockDim == D (128/256)
// =====================================================================
__global__ void k_ln_relu(const float* __restrict__ X, int ldx, int D,
                          const float* __restrict__ g, const float* __restrict__ b,
                          const float* __restrict__ res, int ldr,
                          float* __restrict__ Y, int ldy)
{
  __shared__ float red[256];
  const int row = blockIdx.x, c = threadIdx.x;
  const float v = X[(size_t)row * ldx + c];
  red[c] = v; __syncthreads();
  for (int s = blockDim.x >> 1; s > 0; s >>= 1) {
    if (c < s) red[c] += red[c + s];
    __syncthreads();
  }
  const float mu = red[0] / (float)D;
  __syncthreads();
  const float dlt = v - mu;
  red[c] = dlt * dlt; __syncthreads();
  for (int s = blockDim.x >> 1; s > 0; s >>= 1) {
    if (c < s) red[c] += red[c + s];
    __syncthreads();
  }
  const float var = red[0] / (float)D;
  float y = dlt * rsqrtf(var + LN_EPS) * g[c] + b[c];
  y = fmaxf(y, 0.0f);
  if (res) y += res[(size_t)row * ldr + c];
  Y[(size_t)row * ldy + c] = y;
}

// =====================================================================
// logits[n] = dot(ag[n,:512], Wc) + bc   — one wave per row
// =====================================================================
__global__ void k_rowdot(const float* __restrict__ ag, const float* __restrict__ Wc,
                         const float* __restrict__ bc, float* __restrict__ logits, int n)
{
  const int wid  = threadIdx.x >> 5, lane = threadIdx.x & 31;
  const int row  = blockIdx.x * 8 + wid;
  if (row >= n) return;
  float acc = 0.0f;
  #pragma unroll
  for (int j = 0; j < DCAT / 32; ++j) {
    const int c = lane + 32 * j;
    acc += ag[(size_t)row * DCAT + c] * Wc[c];
  }
  #pragma unroll
  for (int off = 16; off > 0; off >>= 1) acc += __shfl_down(acc, off, 32);
  if (lane == 0) logits[row] = acc + bc[0];
}

// global softmax stats over N logits (1 block)
__global__ void k_smax_stats(const float* __restrict__ logits, int n, float* stats) {
  __shared__ float red[1024];
  const int tid = threadIdx.x;
  float m = -3.4e38f;
  for (int i = tid; i < n; i += 1024) m = fmaxf(m, logits[i]);
  red[tid] = m; __syncthreads();
  for (int s = 512; s > 0; s >>= 1) { if (tid < s) red[tid] = fmaxf(red[tid], red[tid + s]); __syncthreads(); }
  const float mx = red[0]; __syncthreads();
  float sum = 0.0f;
  for (int i = tid; i < n; i += 1024) sum += __expf(logits[i] - mx);
  red[tid] = sum; __syncthreads();
  for (int s = 512; s > 0; s >>= 1) { if (tid < s) red[tid] += red[tid + s]; __syncthreads(); }
  if (tid == 0) { stats[0] = mx; stats[1] = red[0]; }
}

__global__ void k_attn_norm(const float* __restrict__ logits, const float* __restrict__ stats,
                            float* __restrict__ attn, int n) {
  const int i = blockIdx.x * blockDim.x + threadIdx.x;
  if (i < n) attn[i] = __expf(logits[i] - stats[0]) / stats[1];
}

// out1[c] += sum_n attn[n]*X[n,c];  out2[c] += sum_n attn[n]*X[n,c]^2 (optional)
__global__ void k_wreduce(const float* __restrict__ X, int ld, int cols,
                          const float* __restrict__ attn,
                          float* __restrict__ out1, float* __restrict__ out2,
                          int n, int nchunks)
{
  const int c = blockIdx.x * blockDim.x + threadIdx.x;
  if (c >= cols) return;
  const int chunk = (n + nchunks - 1) / nchunks;
  const int n0 = blockIdx.y * chunk;
  const int n1 = min(n0 + chunk, n);
  float a1 = 0.0f, a2 = 0.0f;
  for (int i = n0; i < n1; ++i) {
    const float a = attn[i];
    const float x = X[(size_t)i * ld + c];
    a1 += a * x;
    a2 += a * x * x;
  }
  atomAddF(&out1[c], a1);
  if (out2) atomAddF(&out2[c], a2);
}

// final head: out[k] = [wavg | log(S2-S1^2 + eps)] @ W_head + b_head
__global__ void k_head(const float* __restrict__ wavg, const float* __restrict__ S1,
                       const float* __restrict__ S2, const float* __restrict__ Wh,
                       const float* __restrict__ bh, float* __restrict__ out)
{
  const int t = threadIdx.x;
  if (t >= NCLS) return;
  float acc = bh[t];
  for (int c = 0; c < DCAT; ++c) acc += wavg[c] * Wh[c * NCLS + t];
  for (int j = 0; j < NVP; ++j) {
    const float var = S2[j] - S1[j] * S1[j];
    acc += logf(var + LOG_EPS) * Wh[(DCAT + j) * NCLS + t];
  }
  out[t] = acc;
}

// =====================================================================
// host orchestration
// =====================================================================
static inline void gemm(hipStream_t s, const float* A, int lda, const float* B, int ldb,
                        const float* bias, float* C, int ldc, int M, int K, int Ncols,
                        int act, int cmode) {
  dim3 g((M + GBM - 1) / GBM, (Ncols + GBN - 1) / GBN);
  k_gemm<<<g, 256, 0, s>>>(A, lda, B, ldb, bias, C, ldc, M, K, Ncols, act, cmode);
}
static inline void fillu(hipStream_t s, float* p, uint32_t bits, size_t n) {
  k_fill<<<(unsigned)((n + 255) / 256), 256, 0, s>>>((uint32_t*)p, bits, n);
}

extern "C" void kernel_launch(void* const* d_in, const int* in_sizes, int n_in,
                              void* d_out, int out_size, void* d_ws, size_t ws_size,
                              hipStream_t stream)
{
  const int N = in_sizes[0] / FIN;
  const int E = in_sizes[1] / 2;

  const float* x      = (const float*)d_in[0];
  const int*   eidx   = (const int*)d_in[1];
  const int*   esrc   = eidx;
  const int*   edst   = eidx + E;
  const float* W_fc   = (const float*)d_in[2];
  const float* b_fc   = (const float*)d_in[3];
  const float* tarr   = (const float*)d_in[4];
  const float* W1     = (const float*)d_in[5];
  const float* b1     = (const float*)d_in[6];
  const float* ln1_g  = (const float*)d_in[7];
  const float* ln1_b  = (const float*)d_in[8];
  const float* W2     = (const float*)d_in[9];
  const float* b2     = (const float*)d_in[10];
  const float* ln2_g  = (const float*)d_in[11];
  const float* ln2_b  = (const float*)d_in[12];
  const float* W_phi  = (const float*)d_in[13];
  const float* b_phi  = (const float*)d_in[14];
  const float* Wa     = (const float*)d_in[15];
  const float* ba     = (const float*)d_in[16];
  const float* Wb     = (const float*)d_in[17];
  const float* bb     = (const float*)d_in[18];
  const float* Wc     = (const float*)d_in[19];
  const float* bc     = (const float*)d_in[20];
  const float* W_var  = (const float*)d_in[21];
  const float* b_var  = (const float*)d_in[22];
  const float* W_head = (const float*)d_in[23];
  const float* b_head = (const float*)d_in[24];

  // ---- workspace layout (floats) ----
  float* W = (float*)d_ws;
  size_t o = 0;
  float* feats  = W + o; o += (size_t)N * DCAT;        // [h0|h1|h2|h3]
  float* hp     = W + o; o += (size_t)N * DCAT;
  float* scr    = W + o; o += (size_t)N * 384;         // conv scratch / proj
  float* logits = W + o; o += (size_t)N;
  float* attn   = W + o; o += (size_t)N;
  float* stats  = W + o; o += 8;
  float* wavg   = W + o; o += DCAT;
  float* S1v    = W + o; o += 128;
  float* S2v    = W + o; o += 128;

  float* mxbuf  = scr;                       // [N,128]
  float* denbuf = scr + (size_t)N * HDIM;    // [N,128]
  float* aggbuf = scr + (size_t)2 * N * HDIM;// [N,128]  (later pre-LN h)
  float* zbuf   = scr;                       // [N,256]  (reuses mx+den)
  float* proj   = scr;                       // [N,100]  (after convs done)

  const long long eTot = (long long)E * HDIM;
  const unsigned eGrid = (unsigned)((eTot + 255) / 256);
  const long long nTot = (long long)N * HDIM;
  const unsigned nGrid = (unsigned)((nTot + 255) / 256);

  // ---- h0 = relu(x @ W_fc + b_fc) -> feats cols [0,128) ----
  gemm(stream, x, FIN, W_fc, HDIM, b_fc, feats, DCAT, N, FIN, HDIM, /*relu*/1, 0);

  // ---- 3 GENConv layers ----
  for (int i = 0; i < NLAY; ++i) {
    const int colbase = i * HDIM;
    fillu(stream, mxbuf,  0xFF800000u, (size_t)N * HDIM);  // -inf
    fillu(stream, denbuf, 0u,          (size_t)N * HDIM);
    fillu(stream, aggbuf, 0u,          (size_t)N * HDIM);
    k_edge<<<eGrid, 256, 0, stream>>>(feats, DCAT, colbase, esrc, edst, tarr, i,
                                      mxbuf, denbuf, aggbuf, E, 0);
    k_edge<<<eGrid, 256, 0, stream>>>(feats, DCAT, colbase, esrc, edst, tarr, i,
                                      mxbuf, denbuf, aggbuf, E, 1);
    k_edge<<<eGrid, 256, 0, stream>>>(feats, DCAT, colbase, esrc, edst, tarr, i,
                                      mxbuf, denbuf, aggbuf, E, 2);
    k_add_root<<<nGrid, 256, 0, stream>>>(aggbuf, feats, DCAT, colbase, N);

    // z_pre = (agg+h) @ W1[i] + b1[i]   -> zbuf [N,256]
    gemm(stream, aggbuf, HDIM, W1 + (size_t)i * HDIM * 2 * HDIM, 2 * HDIM,
         b1 + (size_t)i * 2 * HDIM, zbuf, 2 * HDIM, N, HDIM, 2 * HDIM, 0, 0);
    // z = relu(LN1(z_pre)) in place
    k_ln_relu<<<N, 2 * HDIM, 0, stream>>>(zbuf, 2 * HDIM, 2 * HDIM,
                                          ln1_g + (size_t)i * 2 * HDIM,
                                          ln1_b + (size_t)i * 2 * HDIM,
                                          nullptr, 0, zbuf, 2 * HDIM);
    if (i == 0) {
      // h1 = z @ W2[0] + b2[0]  -> feats cols [128,256)
      gemm(stream, zbuf, 2 * HDIM, W2, HDIM, b2, feats + HDIM, DCAT,
           N, 2 * HDIM, HDIM, 0, 0);
    } else {
      // pre-LN conv out -> aggbuf; then h_{i+1} = h_i + relu(LN2(.))
      gemm(stream, zbuf, 2 * HDIM, W2 + (size_t)i * 2 * HDIM * HDIM, HDIM,
           b2 + (size_t)i * HDIM, aggbuf, HDIM, N, 2 * HDIM, HDIM, 0, 0);
      k_ln_relu<<<N, HDIM, 0, stream>>>(aggbuf, HDIM, HDIM,
                                        ln2_g + (size_t)(i - 1) * HDIM,
                                        ln2_b + (size_t)(i - 1) * HDIM,
                                        feats + (size_t)i * HDIM, DCAT,
                                        feats + (size_t)(i + 1) * HDIM, DCAT);
    }
  }

  // ---- hp = relu(feats @ W_phi + b_phi) ----
  gemm(stream, feats, DCAT, W_phi, DCAT, b_phi, hp, DCAT, N, DCAT, DCAT, 1, 0);
  // ---- ag = tanh(hp@Wa+ba) * sigmoid(hp@Wb+bb)  (reuse feats) ----
  gemm(stream, hp, DCAT, Wa, DCAT, ba, feats, DCAT, N, DCAT, DCAT, /*tanh*/2, 0);
  gemm(stream, hp, DCAT, Wb, DCAT, bb, feats, DCAT, N, DCAT, DCAT, /*sigm*/3, /*mul*/1);
  // ---- attn = softmax(ag @ Wc + bc) ----
  k_rowdot<<<(unsigned)((N + 7) / 8), 256, 0, stream>>>(feats, Wc, bc, logits, N);
  k_smax_stats<<<1, 1024, 0, stream>>>(logits, N, stats);
  k_attn_norm<<<(unsigned)((N + 255) / 256), 256, 0, stream>>>(logits, stats, attn, N);
  // ---- proj = hp @ W_var + b_var ----
  gemm(stream, hp, DCAT, W_var, NVP, b_var, proj, NVP, N, DCAT, NVP, 0, 0);
  // ---- weighted reductions ----
  fillu(stream, wavg, 0u, DCAT);
  fillu(stream, S1v,  0u, 128);
  fillu(stream, S2v,  0u, 128);
  {
    dim3 g1((DCAT + 255) / 256, 128);
    k_wreduce<<<g1, 256, 0, stream>>>(hp, DCAT, DCAT, attn, wavg, nullptr, N, 128);
    dim3 g2(1, 128);
    k_wreduce<<<g2, 256, 0, stream>>>(proj, NVP, NVP, attn, S1v, S2v, N, 128);
  }
  // ---- head ----
  k_head<<<1, 32, 0, stream>>>(wavg, S1v, S2v, W_head, b_head, (float*)d_out);
}
